// MoEGPT_56298431316472
// MI455X (gfx1250) — compile-verified, matrix operand order checked by reading, not run on hardware
//
#include <hip/hip_runtime.h>

// ---------------- model constants (match reference config) ----------------
constexpr int Dm  = 768;
constexpr int Lc  = 2;
constexpr int Hc  = 12;
constexpr int Ec  = 8;
constexpr int Sc  = 2048;
constexpr int Bc  = 2;
constexpr int FFc = 4 * Dm;          // 3072
constexpr int Vc  = 32000;
constexpr int Tc  = Bc * Sc;         // 4096 tokens
constexpr int HDc = Dm / Hc;         // 64
constexpr int D3c = 3 * Dm;          // 2304

// ---------------- WMMA types ----------------
typedef __bf16 bf16_t;
typedef bf16_t v16bf __attribute__((ext_vector_type(16)));
typedef bf16_t v2bf  __attribute__((ext_vector_type(2)));
typedef float  v8f   __attribute__((ext_vector_type(8)));

// native f32->bf16 (lets the backend pick packed cvt instructions)
__device__ __forceinline__ unsigned int pack2_bf16(float x, float y) {
    v2bf p;
    p[0] = (bf16_t)x;
    p[1] = (bf16_t)y;
    return __builtin_bit_cast(unsigned int, p);
}
__device__ __forceinline__ unsigned short to_bf16_u16(float x) {
    return __builtin_bit_cast(unsigned short, (bf16_t)x);
}

// ---------------- generic strided GEMM: C = alpha*(A @ B^T) + bias ----------------
// Contract (all call sites satisfy this):
//   A row-major [M,K], contiguous K, row stride asM; M % 128 == 0, K % 32 == 0.
//   B is [N,K]:
//     bsK == 1 : row-major [N,K] with row stride bsN; N % 128 == 0  (weights, K-tiles)
//     bsK != 1 : bsN == 1, i.e. B[k*bsK + n] (K-major = V matrix); N % 4 == 0
//   C row-major [M,N], f32.
// Block: 256 threads = 8 waves, tile 128x128x32.
// Waves: 4 along M x 2 along N; wave tile 32(M) x 64(N) = 2x4 WMMA fragments.
constexpr int BM = 128, BN = 128, BK = 32;
constexpr int LDT = BK + 16;          // ushort row stride 48 -> 96B rows, 16B aligned

__global__ __launch_bounds__(256)
void gemm_bf16_wmma(const float* __restrict__ A, int asM,
                    const float* __restrict__ Bm, int bsN, int bsK,
                    float* __restrict__ C, int ldc,
                    const float* __restrict__ bias,
                    int M, int N, int Kd, float alpha)
{
    __shared__ __align__(16) unsigned short sA[BM][LDT];
    __shared__ __align__(16) unsigned short sB[BN][LDT];

    const int tid  = threadIdx.x;
    const int lane = tid & 31;
    const int wave = tid >> 5;          // 0..7
    const int wm   = wave & 3;          // 4 waves along M
    const int wn   = wave >> 2;         // 2 waves along N
    const int l16  = lane & 15;
    const int half = lane >> 4;         // 0: lanes 0-15, 1: lanes 16-31
    const int kLo  = half * 8;          // K sub-offset per documented bf16 layout

    const int n0 = blockIdx.x * BN;
    const int m0 = blockIdx.y * BM;

    const bool rowMajorB = (bsK == 1);  // uniform per launch

    v8f acc[2][4] = {};

    for (int k0 = 0; k0 < Kd; k0 += BK) {
        // ---- stage A tile (f32 -> bf16), fully coalesced float4 ----
#pragma unroll
        for (int j = 0; j < 4; ++j) {
            int i = tid + j * 256;          // 1024 float4 = 128 x 32 f32
            int r = i >> 3, c4 = (i & 7) * 4;
            const float4 v = *reinterpret_cast<const float4*>(
                A + (long)(m0 + r) * asM + (k0 + c4));
            uint2 pk;
            pk.x = pack2_bf16(v.x, v.y);
            pk.y = pack2_bf16(v.z, v.w);
            *reinterpret_cast<uint2*>(&sA[r][c4]) = pk;
        }
        // ---- stage B tile (f32 -> bf16) ----
        if (rowMajorB) {
#pragma unroll
            for (int j = 0; j < 4; ++j) {
                int i = tid + j * 256;
                int r = i >> 3, c4 = (i & 7) * 4;
                const float4 v = *reinterpret_cast<const float4*>(
                    Bm + (long)(n0 + r) * bsN + (k0 + c4));
                uint2 pk;
                pk.x = pack2_bf16(v.x, v.y);
                pk.y = pack2_bf16(v.z, v.w);
                *reinterpret_cast<uint2*>(&sB[r][c4]) = pk;
            }
        } else {
            // K-major B (attn @ V): contiguous along n within a k row; transpose into LDS.
#pragma unroll
            for (int j = 0; j < 4; ++j) {
                int i = tid + j * 256;          // 1024 float4 = 32 k x 128 n
                int kk = i >> 5, c4 = (i & 31) * 4;
                bool inb = (n0 + c4) < N;
                int cc = inb ? (n0 + c4) : 0;   // clamped, no exec branching
                const float4 v = *reinterpret_cast<const float4*>(
                    Bm + (long)(k0 + kk) * bsK + cc);
                sB[c4 + 0][kk] = to_bf16_u16(inb ? v.x : 0.f);
                sB[c4 + 1][kk] = to_bf16_u16(inb ? v.y : 0.f);
                sB[c4 + 2][kk] = to_bf16_u16(inb ? v.z : 0.f);
                sB[c4 + 3][kk] = to_bf16_u16(inb ? v.w : 0.f);
            }
        }
        // speculative prefetch of the next K-panel (gfx1250 global_prefetch path)
        if (k0 + BK < Kd) {
            int r = tid >> 1, kk = (tid & 1) * 16;
            __builtin_prefetch(A + (long)(m0 + r) * asM + (k0 + BK + kk), 0, 0);
            if (rowMajorB)
                __builtin_prefetch(Bm + (long)(n0 + r) * bsN + (k0 + BK + kk), 0, 0);
            else
                __builtin_prefetch(Bm + (long)(k0 + BK + (tid >> 3)) * bsK + n0 + (tid & 7) * 16, 0, 0);
        }
        __syncthreads();

        // ---- load fragments from LDS per bf16 16x16x32 VGPR layout ----
        v16bf afr[2], bfr[4];
#pragma unroll
        for (int mi = 0; mi < 2; ++mi) {
            int row = wm * 32 + mi * 16 + l16;
            uint4* p = reinterpret_cast<uint4*>(&afr[mi]);
            p[0] = *reinterpret_cast<const uint4*>(&sA[row][kLo]);       // K = kLo..kLo+7
            p[1] = *reinterpret_cast<const uint4*>(&sA[row][kLo + 16]);  // K = kLo+16..kLo+23
        }
#pragma unroll
        for (int ni = 0; ni < 4; ++ni) {
            int col = wn * 64 + ni * 16 + l16;
            uint4* p = reinterpret_cast<uint4*>(&bfr[ni]);
            p[0] = *reinterpret_cast<const uint4*>(&sB[col][kLo]);
            p[1] = *reinterpret_cast<const uint4*>(&sB[col][kLo + 16]);
        }

#pragma unroll
        for (int mi = 0; mi < 2; ++mi)
#pragma unroll
            for (int ni = 0; ni < 4; ++ni)
                acc[mi][ni] = __builtin_amdgcn_wmma_f32_16x16x32_bf16(
                    false, afr[mi], false, bfr[ni], (short)0, acc[mi][ni], false, false);

        __syncthreads();
    }

    // ---- write out: VGPR r -> rows (r, r+8) split across lane halves ----
#pragma unroll
    for (int mi = 0; mi < 2; ++mi)
#pragma unroll
        for (int ni = 0; ni < 4; ++ni) {
            int col = n0 + wn * 64 + ni * 16 + l16;
            float bb = (col < N && bias) ? bias[col] : 0.f;
#pragma unroll
            for (int r = 0; r < 8; ++r) {
                int row = m0 + wm * 32 + mi * 16 + r + half * 8;   // M % 128 == 0: always valid
                if (col < N)
                    C[(long)row * ldc + col] = acc[mi][ni][r] * alpha + bb;
            }
        }
}

// ---------------- embedding gather ----------------
__global__ __launch_bounds__(256)
void embed_k(const int* __restrict__ ids, const float* __restrict__ tok,
             const float* __restrict__ pos, float* __restrict__ out)
{
    int t = blockIdx.x;
    int id = ids[t];
    int s = t % Sc;
    for (int d = threadIdx.x; d < Dm; d += 256)
        out[(long)t * Dm + d] = tok[(long)id * Dm + d] + pos[(long)s * Dm + d];
}

// ---------------- layernorm (one block per row) ----------------
__global__ __launch_bounds__(256)
void layernorm_k(const float* __restrict__ x, const float* __restrict__ w,
                 const float* __restrict__ b, float* __restrict__ y, int Dl)
{
    __shared__ float red[256];
    long row = blockIdx.x;
    const float* xr = x + row * Dl;
    float s = 0.f;
    for (int d = threadIdx.x; d < Dl; d += 256) s += xr[d];
    red[threadIdx.x] = s; __syncthreads();
    for (int off = 128; off > 0; off >>= 1) {
        if (threadIdx.x < off) red[threadIdx.x] += red[threadIdx.x + off];
        __syncthreads();
    }
    float mean = red[0] / Dl;
    __syncthreads();
    float v = 0.f;
    for (int d = threadIdx.x; d < Dl; d += 256) { float t = xr[d] - mean; v += t * t; }
    red[threadIdx.x] = v; __syncthreads();
    for (int off = 128; off > 0; off >>= 1) {
        if (threadIdx.x < off) red[threadIdx.x] += red[threadIdx.x + off];
        __syncthreads();
    }
    float rstd = rsqrtf(red[0] / Dl + 1e-5f);
    float* yr = y + row * Dl;
    for (int d = threadIdx.x; d < Dl; d += 256)
        yr[d] = (xr[d] - mean) * rstd * w[d] + b[d];
}

// ---------------- row softmax (in place) ----------------
__global__ __launch_bounds__(256)
void softmax_rows_k(float* __restrict__ x, int Nl)
{
    __shared__ float red[256];
    long row = blockIdx.x;
    float* xr = x + row * Nl;
    float m = -1e30f;
    for (int i = threadIdx.x; i < Nl; i += 256) m = fmaxf(m, xr[i]);
    red[threadIdx.x] = m; __syncthreads();
    for (int off = 128; off > 0; off >>= 1) {
        if (threadIdx.x < off) red[threadIdx.x] = fmaxf(red[threadIdx.x], red[threadIdx.x + off]);
        __syncthreads();
    }
    m = red[0]; __syncthreads();
    float s = 0.f;
    for (int i = threadIdx.x; i < Nl; i += 256) { float e = __expf(xr[i] - m); xr[i] = e; s += e; }
    red[threadIdx.x] = s; __syncthreads();
    for (int off = 128; off > 0; off >>= 1) {
        if (threadIdx.x < off) red[threadIdx.x] += red[threadIdx.x + off];
        __syncthreads();
    }
    float inv = 1.f / red[0];
    for (int i = threadIdx.x; i < Nl; i += 256) xr[i] *= inv;
}

// ---------------- residual add (in place) ----------------
__global__ __launch_bounds__(256)
void addinto_k(float* __restrict__ dst, const float* __restrict__ src, long n)
{
    long i = blockIdx.x * 256L + threadIdx.x;
    if (i < n) dst[i] += src[i];
}

// ---------------- gelu (tanh approx, jax default) ----------------
__global__ __launch_bounds__(256)
void gelu_k(float* __restrict__ x, long n)
{
    long i = blockIdx.x * 256L + threadIdx.x;
    if (i < n) {
        float v = x[i];
        float c = 0.7978845608028654f * (v + 0.044715f * v * v * v);
        x[i] = 0.5f * v * (1.f + tanhf(c));
    }
}

// ---------------- MoE gate: softmax over E, top-2, normalized combine weights ----------------
__global__ __launch_bounds__(32)
void gate_topk_k(const float* __restrict__ xln, const float* __restrict__ gw,
                 const float* __restrict__ gb, float* __restrict__ cw)
{
    int t = blockIdx.x;
    int lane = threadIdx.x;
    const float* xr = xln + (long)t * Dm;
    float acc[Ec];
#pragma unroll
    for (int e = 0; e < Ec; ++e) acc[e] = 0.f;
    for (int d = lane; d < Dm; d += 32) {
        float xv = xr[d];
#pragma unroll
        for (int e = 0; e < Ec; ++e) acc[e] += xv * gw[e * Dm + d];
    }
#pragma unroll
    for (int e = 0; e < Ec; ++e)
        for (int off = 16; off > 0; off >>= 1)
            acc[e] += __shfl_xor(acc[e], off, 32);
    if (lane == 0) {
        float logit[Ec]; float mx = -1e30f;
#pragma unroll
        for (int e = 0; e < Ec; ++e) { logit[e] = acc[e] + gb[e]; mx = fmaxf(mx, logit[e]); }
        float p[Ec]; float s = 0.f;
#pragma unroll
        for (int e = 0; e < Ec; ++e) { p[e] = __expf(logit[e] - mx); s += p[e]; }
#pragma unroll
        for (int e = 0; e < Ec; ++e) p[e] /= s;
        int i0 = 0;
        for (int e = 1; e < Ec; ++e) if (p[e] > p[i0]) i0 = e;
        int i1 = (i0 == 0) ? 1 : 0;
        for (int e = 0; e < Ec; ++e) if (e != i0 && p[e] > p[i1]) i1 = e;
        float wsum = p[i0] + p[i1];
        for (int e = 0; e < Ec; ++e) cw[(long)t * Ec + e] = 0.f;
        cw[(long)t * Ec + i0] = p[i0] / wsum;
        cw[(long)t * Ec + i1] = p[i1] / wsum;
    }
}

// ---------------- expert combine: x += cw[t,e] * y ----------------
__global__ __launch_bounds__(256)
void combine_k(float* __restrict__ x, const float* __restrict__ y,
               const float* __restrict__ cw, int e, long n)
{
    long i = blockIdx.x * 256L + threadIdx.x;
    if (i < n) {
        long t = i / Dm;
        float w = cw[t * Ec + e];
        x[i] += w * y[i];
    }
}

// ---------------- host orchestration ----------------
extern "C" void kernel_launch(void* const* d_in, const int* in_sizes, int n_in,
                              void* d_out, int out_size, void* d_ws, size_t ws_size,
                              hipStream_t stream)
{
    const int*   ids   = (const int*)  d_in[0];
    const float* tok   = (const float*)d_in[1];
    const float* pos   = (const float*)d_in[2];
    const float* ln1w  = (const float*)d_in[3];
    const float* ln1b  = (const float*)d_in[4];
    const float* qkvw  = (const float*)d_in[5];
    const float* qkvb  = (const float*)d_in[6];
    const float* projw = (const float*)d_in[7];
    const float* projb = (const float*)d_in[8];
    const float* ln2w  = (const float*)d_in[9];
    const float* ln2b  = (const float*)d_in[10];
    const float* gatew = (const float*)d_in[11];
    const float* gateb = (const float*)d_in[12];
    const float* w1    = (const float*)d_in[13];
    const float* b1    = (const float*)d_in[14];
    const float* w2    = (const float*)d_in[15];
    const float* b2    = (const float*)d_in[16];
    const float* lnfw  = (const float*)d_in[17];
    const float* lnfb  = (const float*)d_in[18];
    const float* headw = (const float*)d_in[19];
    const float* headb = (const float*)d_in[20];
    (void)in_sizes; (void)n_in; (void)out_size; (void)ws_size;

    float* ws = (float*)d_ws;
    long off = 0;
    float* xbuf    = ws + off; off += (long)Tc * Dm;
    float* lnbuf   = ws + off; off += (long)Tc * Dm;
    float* qkvbuf  = ws + off; off += (long)Tc * D3c;
    float* scores  = ws + off; off += (long)Sc * Sc;
    float* ctxbuf  = ws + off; off += (long)Tc * Dm;
    float* projbuf = ws + off; off += (long)Tc * Dm;
    float* hbuf    = ws + off; off += (long)Tc * FFc;
    float* ybuf    = ws + off; off += (long)Tc * Dm;
    float* cw      = ws + off; off += (long)Tc * Ec;

    dim3 blk(256);
    const long nTD  = (long)Tc * Dm;
    const long nTFF = (long)Tc * FFc;

    auto gemm = [&](const float* A, int asM,
                    const float* Bm, int bsN, int bsK,
                    float* C, int ldc, const float* bias,
                    int M, int N, int Kd, float alpha) {
        dim3 grid((unsigned)((N + BN - 1) / BN), (unsigned)((M + BM - 1) / BM));
        gemm_bf16_wmma<<<grid, blk, 0, stream>>>(A, asM, Bm, bsN, bsK,
                                                 C, ldc, bias, M, N, Kd, alpha);
    };

    // token + position embedding
    embed_k<<<Tc, blk, 0, stream>>>(ids, tok, pos, xbuf);

    for (int l = 0; l < Lc; ++l) {
        // ---- attention block ----
        layernorm_k<<<Tc, blk, 0, stream>>>(xbuf, ln1w + l * Dm, ln1b + l * Dm, lnbuf, Dm);
        gemm(lnbuf, Dm,
             qkvw + (long)l * D3c * Dm, Dm, 1,
             qkvbuf, D3c, qkvb + l * D3c, Tc, D3c, Dm, 1.f);

        for (int b = 0; b < Bc; ++b) {
            for (int h = 0; h < Hc; ++h) {
                const float* qp = qkvbuf + (long)b * Sc * D3c + h * HDc;
                const float* kp = qp + Dm;
                const float* vp = qp + 2 * Dm;
                // scores = (Q K^T) / sqrt(hd)
                gemm(qp, D3c, kp, D3c, 1, scores, Sc, nullptr, Sc, Sc, HDc, 0.125f);
                softmax_rows_k<<<Sc, blk, 0, stream>>>(scores, Sc);
                // ctx = attn @ V   (B is K-major: element (n,k) at vp[k*D3c + n])
                gemm(scores, Sc, vp, 1, D3c,
                     ctxbuf + (long)b * Sc * Dm + h * HDc, Dm, nullptr, Sc, HDc, Sc, 1.f);
            }
        }
        gemm(ctxbuf, Dm,
             projw + (long)l * Dm * Dm, Dm, 1,
             projbuf, Dm, projb + l * Dm, Tc, Dm, Dm, 1.f);
        addinto_k<<<(unsigned)((nTD + 255) / 256), blk, 0, stream>>>(xbuf, projbuf, nTD);

        // ---- MoE block ----
        layernorm_k<<<Tc, blk, 0, stream>>>(xbuf, ln2w + l * Dm, ln2b + l * Dm, lnbuf, Dm);
        gate_topk_k<<<Tc, 32, 0, stream>>>(lnbuf, gatew + (long)l * Ec * Dm, gateb + l * Ec, cw);
        for (int e = 0; e < Ec; ++e) {
            const float* w1e = w1 + ((long)l * Ec + e) * (long)FFc * Dm;
            const float* b1e = b1 + ((long)l * Ec + e) * FFc;
            const float* w2e = w2 + ((long)l * Ec + e) * (long)Dm * FFc;
            const float* b2e = b2 + ((long)l * Ec + e) * Dm;
            gemm(lnbuf, Dm, w1e, Dm, 1, hbuf, FFc, b1e, Tc, FFc, Dm, 1.f);
            gelu_k<<<(unsigned)((nTFF + 255) / 256), blk, 0, stream>>>(hbuf, nTFF);
            gemm(hbuf, FFc, w2e, FFc, 1, ybuf, Dm, b2e, Tc, Dm, FFc, 1.f);
            combine_k<<<(unsigned)((nTD + 255) / 256), blk, 0, stream>>>(xbuf, ybuf, cw, e, nTD);
        }
    }

    // ---- final LN + LM head ----
    layernorm_k<<<Tc, blk, 0, stream>>>(xbuf, lnfw, lnfb, lnbuf, Dm);
    gemm(lnbuf, Dm, headw, Dm, 1, (float*)d_out, Vc, headb, Tc, Vc, Dm, 1.f);
}
